// BiDirectionalFusionBlock_81475529605665
// MI455X (gfx1250) — compile-verified, matrix operand order checked by reading, not run on hardware
//
#include <hip/hip_runtime.h>

// ---------------------------------------------------------------------------
// Types / helpers
// ---------------------------------------------------------------------------
typedef __attribute__((ext_vector_type(16))) __bf16 v16bf;
typedef __attribute__((ext_vector_type(8)))  float  v8f;

union FragBF { uint4 q[2]; v16bf v; unsigned short s[16]; };

__device__ __forceinline__ float bf2f(unsigned short h) {
    return __uint_as_float(((unsigned)h) << 16);
}
__device__ __forceinline__ unsigned short f2bf(float f) {
    unsigned u = __float_as_uint(f);
    u += 0x7FFFu + ((u >> 16) & 1u);
    return (unsigned short)(u >> 16);
}
__device__ __forceinline__ float silu_f(float x) { return x / (1.f + __expf(-x)); }

// CDNA5 async copy: global -> LDS, 16 bytes per lane, tracked by ASYNCcnt.
__device__ __forceinline__ void async_copy16(unsigned lds_off, const void* gptr) {
    unsigned long long ga = (unsigned long long)gptr;
    asm volatile("global_load_async_to_lds_b128 %0, %1, off"
                 :: "v"(lds_off), "v"(ga) : "memory");
}
__device__ __forceinline__ void wait_async0() {
    asm volatile("s_wait_asynccnt 0x0" ::: "memory");
}

#define Bz   32
#define HSZ  64
#define WSZ  64
#define CSZ  128
#define TSZ  256
#define PC   136          // padded channel stride in LDS (bf16 elements)

// ---------------------------------------------------------------------------
// Bilinear 128x128 -> 64x64 (scale 2, align_corners=False => 2x2 average)
// M: (B,64,128,128) f32 NCHW   ->  Ms: (B,64,64,64) bf16 NHWC
// ---------------------------------------------------------------------------
__global__ void resize_kernel(const float* __restrict__ M, unsigned short* __restrict__ Ms) {
    size_t idx = (size_t)blockIdx.x * blockDim.x + threadIdx.x;
    if (idx >= (size_t)Bz * 64 * 64 * 64) return;
    int c = idx & 63;
    int x = (idx >> 6) & 63;
    int y = (idx >> 12) & 63;
    int b = (int)(idx >> 18);
    const float* s = M + (((size_t)b * 64 + c) * 128 + 2 * y) * 128 + 2 * x;
    float v = 0.25f * (s[0] + s[1] + s[128] + s[129]);
    Ms[idx] = f2bf(v);
}

// ---------------------------------------------------------------------------
// Pack conv weights (OIHW f32) into per-lane WMMA B-fragment order (bf16).
// K ordering: k = tap*128 + ic  (tap = ky*3+kx).  Fragment (kb, nt):
// lane<16 slots: k=kb*32+{0..7,16..23}; lane>=16: +8.
// ---------------------------------------------------------------------------
__global__ void pack_w_kernel(const float* __restrict__ w, unsigned short* __restrict__ dst,
                              int ntaps) {
    int total = (ntaps * 128 / 32) * 8 * 32 * 16;
    int idx = blockIdx.x * blockDim.x + threadIdx.x;
    if (idx >= total) return;
    int j    = idx & 15;
    int lane = (idx >> 4) & 31;
    int nt   = (idx >> 9) & 7;
    int kb   = idx >> 12;
    int half = lane >> 4;
    int klocal = (j < 8) ? (j + (half ? 8 : 0)) : (j - 8 + 16 + (half ? 8 : 0));
    int k  = kb * 32 + klocal;
    int ic = k & 127;
    int tap = k >> 7;
    int oc = nt * 16 + (lane & 15);
    float v;
    if (ntaps == 9) {
        int ky = tap / 3, kx = tap % 3;
        v = w[((size_t)(oc * 128 + ic) * 3 + ky) * 3 + kx];
    } else {
        v = w[oc * 128 + ic];
    }
    dst[idx] = f2bf(v);
}

// ---------------------------------------------------------------------------
// Precompute in_proj t-embed contribution + bias: tcon[b][oc]
// ---------------------------------------------------------------------------
__global__ void tcontrib_kernel(const float* __restrict__ w, const float* __restrict__ bias,
                                const float* __restrict__ temb, float* __restrict__ tcon) {
    int idx = blockIdx.x * blockDim.x + threadIdx.x;
    if (idx >= Bz * CSZ) return;
    int b = idx >> 7, oc = idx & 127;
    float acc = bias[oc];
    for (int k = 0; k < 128; ++k) acc += w[oc * 195 + 67 + k] * temb[b * 128 + k];
    tcon[idx] = acc;
}

__global__ void copy_kernel(const float* __restrict__ s, float* __restrict__ d, int n) {
    int i = blockIdx.x * blockDim.x + threadIdx.x;
    if (i < n) d[i] = s[i];
}

// ---------------------------------------------------------------------------
// Rasterize points -> P (B,64,64,3) f32 : [heat, vx, vy]
// ---------------------------------------------------------------------------
__global__ __launch_bounds__(256)
void raster_kernel(const float* __restrict__ xb, float* __restrict__ P) {
    __shared__ float ptx[TSZ], pty[TSZ], vlx[TSZ], vly[TSZ];
    int b   = blockIdx.x >> 4;
    int px0 = (blockIdx.x & 15) << 8;
    int tid = threadIdx.x;
    float cx = xb[((size_t)b * TSZ + tid) * 2 + 0];
    float cy = xb[((size_t)b * TSZ + tid) * 2 + 1];
    ptx[tid] = (cx + 1.f) * 0.5f * 63.f;
    pty[tid] = (cy + 1.f) * 0.5f * 63.f;
    __syncthreads();
    vlx[tid] = (tid == 0) ? 0.f : ptx[tid] - ptx[tid - 1];
    vly[tid] = (tid == 0) ? 0.f : pty[tid] - pty[tid - 1];
    __syncthreads();
    int pix = px0 + tid;
    float fx = (float)(pix & 63), fy = (float)(pix >> 6);
    const float coef = -0.5f / (1.2f * 1.2f + 1e-8f);
    float wmax = 0.f, sw = 0.f, swx = 0.f, swy = 0.f;
    for (int t = 0; t < TSZ; ++t) {
        float dx = ptx[t] - fx, dy = pty[t] - fy;
        float w = __expf((dx * dx + dy * dy) * coef);
        wmax = fmaxf(wmax, w);
        sw += w; swx += w * vlx[t]; swy += w * vly[t];
    }
    sw = fmaxf(sw, 1e-6f);
    float* o = P + ((size_t)b * 4096 + pix) * 3;
    o[0] = wmax; o[1] = swx / sw; o[2] = swy / sw;
}

// ---------------------------------------------------------------------------
// in_proj 1x1: [Ms(64) | P(3)] @ W(:,0:67) + tcon  -> S (B,64,64,128) bf16
// one block per (row, batch)
// ---------------------------------------------------------------------------
__global__ __launch_bounds__(256)
void in_proj_kernel(const unsigned short* __restrict__ Ms, const float* __restrict__ P,
                    const float* __restrict__ w, const float* __restrict__ tcon,
                    unsigned short* __restrict__ S) {
    __shared__ float wl[128 * 67];
    __shared__ float inl[64][68];
    int y = blockIdx.x, b = blockIdx.y, tid = threadIdx.x;
    for (int i = tid; i < 128 * 67; i += 256) {
        int oc = i / 67, ic = i % 67;
        wl[i] = w[oc * 195 + ic];
    }
    for (int i = tid; i < 64 * 64; i += 256) {
        int px = i >> 6, c = i & 63;
        inl[px][c] = bf2f(Ms[(((size_t)b * 64 + y) * 64 + px) * 64 + c]);
    }
    for (int i = tid; i < 64 * 3; i += 256) {
        int px = i / 3, c = i % 3;
        inl[px][64 + c] = P[(((size_t)b * 64 + y) * 64 + px) * 3 + c];
    }
    __syncthreads();
    int oc = tid & 127;
    int ph = tid >> 7;
    float acc[32];
    float base = tcon[b * 128 + oc];
    #pragma unroll
    for (int p = 0; p < 32; ++p) acc[p] = base;
    for (int ic = 0; ic < 67; ++ic) {
        float wv = wl[oc * 67 + ic];
        #pragma unroll
        for (int p = 0; p < 32; ++p) acc[p] += wv * inl[ph * 32 + p][ic];
    }
    for (int p = 0; p < 32; ++p) {
        int px = ph * 32 + p;
        S[(((size_t)b * 64 + y) * 64 + px) * 128 + oc] = f2bf(acc[p]);
    }
}

// ---------------------------------------------------------------------------
// WMMA conv (3x3 SAME or 1x1), NHWC bf16, optional fused GN-affine+SiLU on
// input load and fused GroupNorm statistics on output.
// One workgroup = one (batch,row): 64 px x 128 oc. 8 waves, each 16px x 64oc.
// When no pre-transform is needed the input tile is staged with CDNA5
// GLOBAL_LOAD_ASYNC_TO_LDS_B128 (ASYNCcnt), bypassing VGPRs entirely.
// ---------------------------------------------------------------------------
template <int NTAPS>
__global__ __launch_bounds__(256)
void conv_wmma_kernel(const unsigned short* __restrict__ in,
                      unsigned short* __restrict__ out,
                      const unsigned short* __restrict__ wfrag,
                      const float* __restrict__ bias,
                      const float* __restrict__ pre_scale,
                      const float* __restrict__ pre_shift,
                      int pre_silu,
                      float* __restrict__ stats) {
    constexpr int ROWS = (NTAPS == 9) ? 3 : 1;
    constexpr int KB   = NTAPS * 128 / 32;
    __shared__ __align__(16) unsigned short tile[ROWS * 64 * PC];
    __shared__ float scl[128];
    __shared__ float shf[128];

    const int y = blockIdx.x, b = blockIdx.y;
    const int tid = threadIdx.x;
    const int lane = tid & 31;
    const int wave = tid >> 5;
    const bool has_pre = (pre_scale != nullptr);

    // stage per-(b,c) GN affine into LDS (pre path only)
    if (has_pre && tid < 128) {
        scl[tid] = pre_scale[b * 128 + tid];
        shf[tid] = pre_shift[b * 128 + tid];
    }
    __syncthreads();

    // ---- stage input (ROWS rows x 64 px x 128 ch) as bf16 into LDS ----
    const int nchunks = ROWS * 64 * 16;                   // 8 bf16 per chunk
    for (int ch = tid; ch < nchunks; ch += 256) {
        int ry  = ch >> 10;                               // /1024
        int rem = ch & 1023;
        int x   = rem >> 4;
        int c   = (rem & 15) << 3;
        int gy  = y + ry - ((NTAPS == 9) ? 1 : 0);
        unsigned short* dstp = &tile[(ry * 64 + x) * PC + c];
        if ((unsigned)gy < 64u) {
            const unsigned short* src = in + (((size_t)b * 64 + gy) * 64 + x) * 128 + c;
            if (!has_pre) {
                // async cache->LDS copy, no VGPR round-trip
                async_copy16((unsigned)(size_t)dstp, src);
            } else {
                unsigned short v[8];
                *(uint4*)v = *(const uint4*)src;
                #pragma unroll
                for (int j = 0; j < 8; ++j) {
                    float f = bf2f(v[j]);
                    f = f * scl[c + j] + shf[c + j];
                    if (pre_silu) f = silu_f(f);
                    v[j] = f2bf(f);
                }
                *(uint4*)dstp = *(uint4*)v;
            }
        } else {
            uint4 z = make_uint4(0, 0, 0, 0);
            *(uint4*)dstp = z;
        }
    }
    if (!has_pre) wait_async0();
    __syncthreads();

    const int mt = wave & 3;        // m-tile (16 px)
    const int nh = wave >> 2;       // oc half (64 oc)
    const int x0 = mt * 16;
    const int m  = lane & 15;
    const int half = lane >> 4;

    v8f acc[4];
    #pragma unroll
    for (int t = 0; t < 4; ++t)
        #pragma unroll
        for (int j = 0; j < 8; ++j) acc[t][j] = 0.f;

    for (int kb = 0; kb < KB; ++kb) {
        const int tap = kb >> 2;
        const int icb = (kb & 3) << 5;
        const int ry  = (NTAPS == 9) ? (tap / 3) : 0;
        const int dx  = (NTAPS == 9) ? ((tap % 3) - 1) : 0;
        const int xx  = x0 + m + dx;

        // prefetch next K-block's weight fragments into cache
        if (kb + 1 < KB)
            __builtin_prefetch(wfrag + (((size_t)(kb + 1) * 8 + nh * 4) * 32 + lane) * 16, 0, 0);

        FragBF a;
        if ((unsigned)xx < 64u) {
            const unsigned short* p = &tile[(ry * 64 + xx) * PC + icb + (half ? 8 : 0)];
            a.q[0] = *(const uint4*)p;          // k-local +0..7  (or +8..15)
            a.q[1] = *(const uint4*)(p + 16);   // k-local +16..23 (or +24..31)
        } else {
            a.q[0] = make_uint4(0, 0, 0, 0);
            a.q[1] = make_uint4(0, 0, 0, 0);
        }
        #pragma unroll
        for (int t = 0; t < 4; ++t) {
            const int nt = nh * 4 + t;
            FragBF bm;
            const unsigned short* wp = wfrag + (((size_t)kb * 8 + nt) * 32 + lane) * 16;
            bm.q[0] = *(const uint4*)wp;
            bm.q[1] = *(const uint4*)(wp + 8);
            acc[t] = __builtin_amdgcn_wmma_f32_16x16x32_bf16(
                false, a.v, false, bm.v, (short)0, acc[t], false, false);
        }
    }

    // ---- epilogue: bias, bf16 store, fused GN statistics ----
    const int nlane = lane & 15;
    #pragma unroll
    for (int t = 0; t < 4; ++t) {
        const int nbase = nh * 64 + t * 16;
        const int cch = nbase + nlane;
        const float bv = bias[cch];
        float bsum = 0.f, bsq = 0.f;
        #pragma unroll
        for (int j = 0; j < 8; ++j) {
            float f = acc[t][j] + bv;
            bsum += f; bsq += f * f;
            int px = x0 + (half ? 8 + j : j);
            out[(((size_t)b * 64 + y) * 64 + px) * 128 + cch] = f2bf(f);
        }
        if (stats) {
            #pragma unroll
            for (int mk = 16; mk >= 1; mk >>= 1) {
                bsum += __shfl_xor(bsum, mk, 32);
                bsq  += __shfl_xor(bsq, mk, 32);
            }
            if (lane == 0) {
                int g = nbase >> 4;
                atomicAdd(&stats[(b * 8 + g) * 2 + 0], bsum);
                atomicAdd(&stats[(b * 8 + g) * 2 + 1], bsq);
            }
        }
    }
}

// ---------------------------------------------------------------------------
// GN finalize: per-(b,c) scale/shift from group sums + gamma/beta
// ---------------------------------------------------------------------------
__global__ void gn_finalize_kernel(const float* __restrict__ stats,
                                   const float* __restrict__ gamma,
                                   const float* __restrict__ beta,
                                   float* __restrict__ scale, float* __restrict__ shift) {
    int idx = blockIdx.x * blockDim.x + threadIdx.x;
    if (idx >= Bz * CSZ) return;
    int b = idx >> 7, c = idx & 127;
    int g = c >> 4;
    const float n = 65536.f;                      // (C/G)*H*W = 16*4096
    float s = stats[(b * 8 + g) * 2 + 0];
    float sq = stats[(b * 8 + g) * 2 + 1];
    float mu = s / n;
    float var = sq / n - mu * mu;
    float inv = rsqrtf(var + 1e-5f);
    float sc = gamma[c] * inv;
    scale[idx] = sc;
    shift[idx] = beta[c] - mu * sc;
}

// ---------------------------------------------------------------------------
// Residual: S = silu(S + gn2(o2))
// ---------------------------------------------------------------------------
__global__ __launch_bounds__(256)
void residual_kernel(unsigned short* __restrict__ S, const unsigned short* __restrict__ o2,
                     const float* __restrict__ scale, const float* __restrict__ shift) {
    size_t idx = (size_t)blockIdx.x * blockDim.x + threadIdx.x;
    int c = (int)(idx & 127);
    int b = (int)(idx >> 19);                     // 4096*128 = 2^19
    float s = bf2f(S[idx]);
    float h = bf2f(o2[idx]) * scale[b * 128 + c] + shift[b * 128 + c];
    S[idx] = f2bf(silu_f(s + h));
}

// ---------------------------------------------------------------------------
// Fused grid_sample + 3-layer MLP + point update. 8 tokens / 128 threads.
// ---------------------------------------------------------------------------
__global__ __launch_bounds__(128)
void corrector_kernel(const unsigned short* __restrict__ Sout,
                      const float* __restrict__ xin, const float* __restrict__ temb,
                      const float* __restrict__ w1, const float* __restrict__ b1,
                      const float* __restrict__ w2, const float* __restrict__ b2,
                      const float* __restrict__ w3, const float* __restrict__ b3,
                      float* __restrict__ xout) {
    __shared__ float feat[8][128];
    __shared__ float h1[8][128];
    __shared__ float tl[128];
    __shared__ float cx0[8], cy0[8], cwx[8], cwy[8];
    int blk = blockIdx.x;
    int b = blk >> 5;
    int tok0 = (blk & 31) * 8;
    int tid = threadIdx.x;
    tl[tid] = temb[b * 128 + tid];
    if (tid < 8) {
        int t = tok0 + tid;
        float nx = xin[((size_t)b * TSZ + t) * 2 + 0];
        float ny = xin[((size_t)b * TSZ + t) * 2 + 1];
        float gx = (nx + 1.f) * 32.f - 0.5f;      // ((x+1)*W - 1)/2, W=64
        float gy = (ny + 1.f) * 32.f - 0.5f;
        float fx0 = floorf(gx), fy0 = floorf(gy);
        cx0[tid] = fx0; cy0[tid] = fy0;
        cwx[tid] = gx - fx0; cwy[tid] = gy - fy0;
    }
    __syncthreads();
    for (int tk = 0; tk < 8; ++tk) {
        float xf0 = cx0[tk], yf0 = cy0[tk], wx = cwx[tk], wy = cwy[tk];
        float acc = 0.f;
        #pragma unroll
        for (int cn = 0; cn < 4; ++cn) {
            float xf = xf0 + (cn & 1);
            float yf = yf0 + (cn >> 1);
            float wg = ((cn & 1) ? wx : 1.f - wx) * ((cn >> 1) ? wy : 1.f - wy);
            if (xf >= 0.f && xf <= 63.f && yf >= 0.f && yf <= 63.f) {
                int xi = (int)xf, yi = (int)yf;
                acc += wg * bf2f(Sout[(((size_t)b * 64 + yi) * 64 + xi) * 128 + tid]);
            }
        }
        feat[tk][tid] = acc;
    }
    __syncthreads();
    // layer 1
    {
        float acc[8];
        float bv = b1[tid];
        float tpart = 0.f;
        for (int k = 0; k < 128; ++k) tpart += tl[k] * w1[(128 + k) * 128 + tid];
        #pragma unroll
        for (int tk = 0; tk < 8; ++tk) acc[tk] = bv + tpart;
        for (int k = 0; k < 128; ++k) {
            float wv = w1[k * 128 + tid];
            #pragma unroll
            for (int tk = 0; tk < 8; ++tk) acc[tk] += feat[tk][k] * wv;
        }
        #pragma unroll
        for (int tk = 0; tk < 8; ++tk) h1[tk][tid] = silu_f(acc[tk]);
    }
    __syncthreads();
    // layer 2 (write back into feat)
    {
        float acc[8];
        float bv = b2[tid];
        #pragma unroll
        for (int tk = 0; tk < 8; ++tk) acc[tk] = bv;
        for (int k = 0; k < 128; ++k) {
            float wv = w2[k * 128 + tid];
            #pragma unroll
            for (int tk = 0; tk < 8; ++tk) acc[tk] += h1[tk][k] * wv;
        }
        __syncthreads();
        #pragma unroll
        for (int tk = 0; tk < 8; ++tk) feat[tk][tid] = silu_f(acc[tk]);
    }
    __syncthreads();
    // layer 3 + update
    if (tid < 16) {
        int tk = tid >> 1, d = tid & 1;
        float acc = b3[d];
        for (int k = 0; k < 128; ++k) acc += feat[tk][k] * w3[k * 2 + d];
        int t = tok0 + tk;
        float xo = xin[((size_t)b * TSZ + t) * 2 + d] + 0.2f * acc;
        xo = fminf(fmaxf(xo, -1.f), 1.f);
        xout[((size_t)b * TSZ + t) * 2 + d] = xo;
    }
}

// ---------------------------------------------------------------------------
// Host launcher
// ---------------------------------------------------------------------------
extern "C" void kernel_launch(void* const* d_in, const int* in_sizes, int n_in,
                              void* d_out, int out_size, void* d_ws, size_t ws_size,
                              hipStream_t stream) {
    (void)in_sizes; (void)n_in; (void)out_size; (void)ws_size;
    const float* M         = (const float*)d_in[0];
    const float* x0        = (const float*)d_in[1];
    const float* temb      = (const float*)d_in[2];
    const float* in_proj_w = (const float*)d_in[3];
    const float* in_proj_b = (const float*)d_in[4];
    const float* rb_c1w    = (const float*)d_in[5];
    const float* rb_c1b    = (const float*)d_in[6];
    const float* rb_g1w    = (const float*)d_in[7];
    const float* rb_g1b    = (const float*)d_in[8];
    const float* rb_c2w    = (const float*)d_in[9];
    const float* rb_c2b    = (const float*)d_in[10];
    const float* rb_g2w    = (const float*)d_in[11];
    const float* rb_g2b    = (const float*)d_in[12];
    const float* out_w     = (const float*)d_in[13];
    const float* out_b     = (const float*)d_in[14];
    const float* w1        = (const float*)d_in[15];
    const float* b1        = (const float*)d_in[16];
    const float* w2        = (const float*)d_in[17];
    const float* b2        = (const float*)d_in[18];
    const float* w3        = (const float*)d_in[19];
    const float* b3        = (const float*)d_in[20];

    char* wsp = (char*)d_ws;
    auto alloc = [&](size_t bytes) -> char* {
        char* p = wsp;
        wsp += (bytes + 255) & ~(size_t)255;
        return p;
    };
    const size_t act = (size_t)Bz * 4096 * 128 * 2;   // bf16 activation map
    unsigned short* Ms   = (unsigned short*)alloc((size_t)Bz * 4096 * 64 * 2);
    float*          P    = (float*)alloc((size_t)Bz * 4096 * 3 * 4);
    unsigned short* S    = (unsigned short*)alloc(act);
    unsigned short* o1   = (unsigned short*)alloc(act);
    unsigned short* o2   = (unsigned short*)alloc(act);
    unsigned short* wf_c1[3], *wf_c2[3];
    for (int i = 0; i < 3; ++i) wf_c1[i] = (unsigned short*)alloc(147456 * 2);
    for (int i = 0; i < 3; ++i) wf_c2[i] = (unsigned short*)alloc(147456 * 2);
    unsigned short* wf_out = (unsigned short*)alloc(16384 * 2);
    float* tcon  = (float*)alloc(Bz * 128 * 4);
    float* stats = (float*)alloc(Bz * 8 * 2 * 4);
    float* sc1   = (float*)alloc(Bz * 128 * 4);
    float* sh1   = (float*)alloc(Bz * 128 * 4);
    float* sc2   = (float*)alloc(Bz * 128 * 4);
    float* sh2   = (float*)alloc(Bz * 128 * 4);
    float* xbuf  = (float*)alloc(Bz * TSZ * 2 * 4);
    unsigned short* Sout = o1;   // out-proj reuses o1 (free after block loop)

    // one-time prep
    resize_kernel<<<(Bz * 64 * 64 * 64 + 255) / 256, 256, 0, stream>>>(M, Ms);
    for (int i = 0; i < 3; ++i) {
        pack_w_kernel<<<576, 256, 0, stream>>>(rb_c1w + (size_t)i * 147456, wf_c1[i], 9);
        pack_w_kernel<<<576, 256, 0, stream>>>(rb_c2w + (size_t)i * 147456, wf_c2[i], 9);
    }
    pack_w_kernel<<<64, 256, 0, stream>>>(out_w, wf_out, 1);
    tcontrib_kernel<<<16, 256, 0, stream>>>(in_proj_w, in_proj_b, temb, tcon);
    copy_kernel<<<64, 256, 0, stream>>>(x0, xbuf, Bz * TSZ * 2);

    for (int it = 0; it < 3; ++it) {
        raster_kernel<<<Bz * 16, 256, 0, stream>>>(xbuf, P);
        in_proj_kernel<<<dim3(64, Bz), 256, 0, stream>>>(Ms, P, in_proj_w, tcon, S);
        for (int blkI = 0; blkI < 3; ++blkI) {
            hipMemsetAsync(stats, 0, Bz * 8 * 2 * 4, stream);
            conv_wmma_kernel<9><<<dim3(64, Bz), 256, 0, stream>>>(
                S, o1, wf_c1[blkI], rb_c1b + blkI * 128, nullptr, nullptr, 0, stats);
            gn_finalize_kernel<<<16, 256, 0, stream>>>(
                stats, rb_g1w + blkI * 128, rb_g1b + blkI * 128, sc1, sh1);
            hipMemsetAsync(stats, 0, Bz * 8 * 2 * 4, stream);
            conv_wmma_kernel<9><<<dim3(64, Bz), 256, 0, stream>>>(
                o1, o2, wf_c2[blkI], rb_c2b + blkI * 128, sc1, sh1, 1, stats);
            gn_finalize_kernel<<<16, 256, 0, stream>>>(
                stats, rb_g2w + blkI * 128, rb_g2b + blkI * 128, sc2, sh2);
            residual_kernel<<<65536, 256, 0, stream>>>(S, o2, sc2, sh2);
        }
        conv_wmma_kernel<1><<<dim3(64, Bz), 256, 0, stream>>>(
            S, Sout, wf_out, out_b, nullptr, nullptr, 0, nullptr);
        float* xo = (it == 2) ? (float*)d_out : xbuf;
        corrector_kernel<<<Bz * 32, 128, 0, stream>>>(
            Sout, xbuf, temb, w1, b1, w2, b2, w3, b3, xo);
    }
}
